// InferenceNet_9139690405960
// MI455X (gfx1250) — matrix operand order repaired and checked by reading, not compiled, not run on hardware
//
#include <hip/hip_runtime.h>
#include <hip/hip_bf16.h>

#define BATCH   256
#define SEQ     512
#define IN_DIM  65
#define IN_PAD  96      // pad K for 3 x 32 WMMA K-tiles
#define HIDDEN  256
#define G3      768     // 3*HIDDEN gate columns
#define RULE    64
#define H_INIT  0.1f

// fragment-workspace geometry (bf16 elements)
#define FRAG_ELEMS 512                  // 32 lanes x 16 halves
#define WHH_FRAGS  (48 * 8)             // 48 N-tiles x 8 K-tiles
#define WIH_FRAGS  (48 * 3)             // 48 N-tiles x 3 K-tiles (K padded 65->96)
#define WOUT_FRAGS (4 * 8)              // 4 N-tiles x 8 K-tiles
#define WHH_E  (WHH_FRAGS * FRAG_ELEMS)
#define WIH_E  (WIH_FRAGS * FRAG_ELEMS)
#define WOUT_E (WOUT_FRAGS * FRAG_ELEMS)
#define TOTAL_E (WHH_E + WIH_E + WOUT_E)        // 286,720 bf16 = 573,440 bytes

typedef __attribute__((ext_vector_type(16))) __bf16 v16bf;
typedef __attribute__((ext_vector_type(8)))  float  v8f;

// ---- WMMA fragment loaders (contiguous k = 16*(lane/16) + s in A and B, wave32) ----

// A-matrix 16x32 bf16 fragment from LDS (row-major, leading dim = ld).
__device__ inline v16bf load_a_frag_lds(const __bf16* S, int ld, int m, int hi8, int kbase) {
  v16bf a;
  const __bf16* p0 = S + m * ld + kbase + 2 * hi8;
  const __bf16* p1 = p0 + 8;
#pragma unroll
  for (int i = 0; i < 8; ++i) { a[i] = p0[i]; a[8 + i] = p1[i]; }
  return a;
}

// A-matrix 16x32 fragment gathered directly from global fp32 (row-major), cvt inline.
__device__ inline v16bf load_a_frag_gf32(const float* __restrict__ S, int ld, size_t row,
                                         int hi8, int kbase) {
  v16bf a;
  const float* p0 = S + row * ld + kbase + 2 * hi8;
  const float* p1 = p0 + 8;
#pragma unroll
  for (int i = 0; i < 8; ++i) { a[i] = (__bf16)p0[i]; a[8 + i] = (__bf16)p1[i]; }
  return a;
}

// B-matrix 32x16 bf16 fragment streamed from global fp32 weights (fallback path).
__device__ inline v16bf load_b_frag_gf32(const float* __restrict__ W, int ldk, int col, int kbase) {
  v16bf b;
  const float* p = W + (size_t)col * ldk + kbase;
#pragma unroll
  for (int i = 0; i < 16; ++i) b[i] = (__bf16)p[i];
  return b;
}

// Same but with K bound (zero pad) for W_ih (fallback path)
__device__ inline v16bf load_b_frag_gf32_pad(const float* __restrict__ W, int ldk, int kmax,
                                             int col, int kbase) {
  v16bf b;
#pragma unroll
  for (int i = 0; i < 16; ++i) {
    int k = kbase + i;
    b[i] = (k < kmax) ? (__bf16)W[(size_t)col * ldk + k] : (__bf16)0.0f;
  }
  return b;
}

// B fragment from pre-converted workspace: frag f, lane l -> 32 contiguous bytes.
__device__ inline v16bf load_b_frag_ws(const __bf16* __restrict__ F, int frag, int lane) {
  return *(const v16bf*)(F + (size_t)frag * FRAG_ELEMS + lane * 16);
}

// ---------------- one-time weight pre-convert into fragment-ready bf16 layout ----------------
__global__ __launch_bounds__(512) void preconvert_kernel(
    const float* __restrict__ W_ih, const float* __restrict__ W_hh,
    const float* __restrict__ W_out, __bf16* __restrict__ ws) {
  int e = blockIdx.x * 512 + threadIdx.x;
  if (e >= TOTAL_E) return;
  int r    = e & (FRAG_ELEMS - 1);
  int lane = r >> 4;
  int i    = r & 15;
  int colL = lane & 15;
  int kL   = ((lane >= 16) ? 16 : 0) + i;
  float v;
  if (e < WHH_E) {
    int f = e >> 9;                      // frag = nt*8 + kt
    int nt = f >> 3, kt = f & 7;
    v = W_hh[(size_t)(nt * 16 + colL) * HIDDEN + kt * 32 + kL];
  } else if (e < WHH_E + WIH_E) {
    int f = (e - WHH_E) >> 9;            // frag = nt*3 + kt
    int nt = f / 3, kt = f - nt * 3;
    int k = kt * 32 + kL;
    v = (k < IN_DIM) ? W_ih[(size_t)(nt * 16 + colL) * IN_DIM + k] : 0.0f;
  } else {
    int f = (e - WHH_E - WIH_E) >> 9;    // frag = nt*8 + kt
    int nt = f >> 3, kt = f & 7;
    v = W_out[(size_t)(nt * 16 + colL) * HIDDEN + kt * 32 + kL];
  }
  ws[e] = (__bf16)v;
}

// ---------------- Fused GRU recurrence: one workgroup per 16-row batch tile ----------------
// 512 threads = 16 waves; wave w owns gate-column tiles 3w..3w+2 (48 tiles = 768 cols).
// h resident in LDS as bf16; weight fragments streamed from L2 every step (bf16 ws frags,
// or f32 fallback). K-tile loops kept as real loops so LICM cannot hoist the full weight
// set into registers (it cannot fit: 540 KB bf16 > WGP register file) and spill.
template <bool USE_WS>
__global__ __launch_bounds__(512) void gru_relu_kernel(
    const float* __restrict__ stim, const float* __restrict__ W_ih,
    const float* __restrict__ W_hh, const float* __restrict__ b_ih,
    const float* __restrict__ b_hh, const __bf16* __restrict__ whh_fr,
    const __bf16* __restrict__ wih_fr, float* __restrict__ hid_out) {
  __shared__ float  pre[16 * G3];        // gi+gh for r,z cols; gi_n for n cols   (48 KB)
  __shared__ float  ghn[16 * HIDDEN];    // gh_n separately (needed for r * gh_n) (16 KB)
  __shared__ __bf16 h_bf[16 * HIDDEN];   // resident hidden state                  (8 KB)
  __shared__ __bf16 x_bf[16 * IN_PAD];   // staged stimulus, zero-padded           (3 KB)
  __shared__ float  bih_s[G3];
  __shared__ float  bhh_s[G3];

  const int tid  = threadIdx.x;
  const int lane = tid & 31;
  const int wave = tid >> 5;
  const int b0   = blockIdx.x * 16;

  const int m      = lane & 15;
  const int hi8    = (lane >> 4) * 8;
  const int koff16 = (lane >= 16) ? 16 : 0;
  int col[3];
#pragma unroll
  for (int j = 0; j < 3; ++j) col[j] = (wave * 3 + j) * 16 + (lane & 15);

  for (int i = tid; i < G3; i += 512) { bih_s[i] = b_ih[i]; bhh_s[i] = b_hh[i]; }
  for (int i = tid; i < 16 * HIDDEN; i += 512) h_bf[i] = (__bf16)H_INIT;
  __syncthreads();

  for (int t = 0; t < SEQ; ++t) {
    // stage x_t (pad cols 65..95 with zero) + prefetch next step's stimulus
    for (int i = tid; i < 16 * IN_PAD; i += 512) {
      int r = i / IN_PAD, c = i - r * IN_PAD;
      float v = 0.0f;
      if (c < IN_DIM) {
        const float* sp = stim + ((size_t)(b0 + r) * SEQ + t) * IN_DIM + c;
        v = *sp;
        if (t + 1 < SEQ) __builtin_prefetch(sp + IN_DIM, 0, 1);
      }
      x_bf[i] = (__bf16)v;
    }
    __syncthreads();

    // gh = h @ W_hh^T  (K = 256, 8 K-tiles), gi = x @ W_ih^T (K = 96 padded, 3 K-tiles)
    v8f acc[3]  = {};
    v8f accx[3] = {};
#pragma unroll 2
    for (int kt = 0; kt < 8; ++kt) {
      v16bf a = load_a_frag_lds(h_bf, HIDDEN, m, hi8, kt * 32);
#pragma unroll
      for (int j = 0; j < 3; ++j) {
        v16bf bf;
        if constexpr (USE_WS) bf = load_b_frag_ws(whh_fr, (wave * 3 + j) * 8 + kt, lane);
        else                  bf = load_b_frag_gf32(W_hh, HIDDEN, col[j], kt * 32 + koff16);
        acc[j] = __builtin_amdgcn_wmma_f32_16x16x32_bf16(false, a, false, bf,
                                                         (short)0, acc[j], false, false);
      }
    }
#pragma unroll 1
    for (int kt = 0; kt < 3; ++kt) {
      v16bf a = load_a_frag_lds(x_bf, IN_PAD, m, hi8, kt * 32);
#pragma unroll
      for (int j = 0; j < 3; ++j) {
        v16bf bf;
        if constexpr (USE_WS) bf = load_b_frag_ws(wih_fr, (wave * 3 + j) * 3 + kt, lane);
        else                  bf = load_b_frag_gf32_pad(W_ih, IN_DIM, IN_DIM, col[j],
                                                        kt * 32 + koff16);
        accx[j] = __builtin_amdgcn_wmma_f32_16x16x32_bf16(false, a, false, bf,
                                                          (short)0, accx[j], false, false);
      }
    }

    // scatter accumulators (C layout: M = v + 8*(lane>=16), N = lane&15 within tile)
#pragma unroll
    for (int j = 0; j < 3; ++j) {
      int c = col[j];
#pragma unroll
      for (int vv = 0; vv < 8; ++vv) {
        int mm = vv + ((lane >> 4) << 3);
        if (c < 2 * HIDDEN) {
          pre[mm * G3 + c] = acc[j][vv] + accx[j][vv];          // r,z: gi+gh combined
        } else {
          pre[mm * G3 + c] = accx[j][vv];                       // gi_n
          ghn[mm * HIDDEN + (c - 2 * HIDDEN)] = acc[j][vv];     // gh_n
        }
      }
    }
    __syncthreads();

    // fp32 gate combine + h update + stream hidden state to HBM
    for (int i = tid; i < 16 * HIDDEN; i += 512) {
      int mm = i >> 8, j = i & 255;
      float rp = pre[mm * G3 + j] + bih_s[j] + bhh_s[j];
      float zp = pre[mm * G3 + HIDDEN + j] + bih_s[HIDDEN + j] + bhh_s[HIDDEN + j];
      float r  = 1.0f / (1.0f + __expf(-rp));
      float z  = 1.0f / (1.0f + __expf(-zp));
      float gh_n = ghn[i] + bhh_s[2 * HIDDEN + j];
      float gi_n = pre[mm * G3 + 2 * HIDDEN + j] + bih_s[2 * HIDDEN + j];
      float nv   = fmaxf(0.0f, gi_n + r * gh_n);
      float hold = (float)h_bf[i];
      float hnew = (1.0f - z) * nv + z * hold;
      h_bf[i] = (__bf16)hnew;
      hid_out[((size_t)(b0 + mm) * SEQ + t) * HIDDEN + j] = hnew;
    }
    __syncthreads();
  }
}

// ---------------- Output head: out = relu(hid @ W_out^T + b_out), [B*T,256]@[256,64] ----------------
// 256 threads = 8 waves; each wave owns one 16-row M-tile and all 4 N-tiles.
// A gathered directly from global fp32 hid; B from pre-converted ws frags (or LDS-staged fallback).
template <bool USE_WS>
__global__ __launch_bounds__(256) void head_kernel(
    const float* __restrict__ hid, const float* __restrict__ W_out,
    const float* __restrict__ b_out, const __bf16* __restrict__ wout_fr,
    float* __restrict__ out) {
  const int tid  = threadIdx.x;
  const int lane = tid & 31;
  const int wave = tid >> 5;

  const size_t mt  = (size_t)blockIdx.x * 8 + wave;   // M-tile index over B*T/16
  const int m      = lane & 15;
  const int hi8    = (lane >> 4) * 8;
  const int koff16 = (lane >= 16) ? 16 : 0;

  __shared__ __bf16 wout_bf[RULE * HIDDEN];           // used by fallback path only
  if constexpr (!USE_WS) {
    for (int i = tid; i < RULE * HIDDEN; i += 256) wout_bf[i] = (__bf16)W_out[i];
    __syncthreads();
  }

  v8f acc[4] = {};
#pragma unroll 2
  for (int kt = 0; kt < 8; ++kt) {
    v16bf a = load_a_frag_gf32(hid, HIDDEN, mt * 16 + m, hi8, kt * 32);
#pragma unroll
    for (int nt = 0; nt < 4; ++nt) {
      v16bf bf;
      if constexpr (USE_WS) bf = load_b_frag_ws(wout_fr, nt * 8 + kt, lane);
      else {
        int c = nt * 16 + (lane & 15);
        const __bf16* p = wout_bf + c * HIDDEN + kt * 32 + koff16;
#pragma unroll
        for (int i = 0; i < 16; ++i) bf[i] = p[i];
      }
      acc[nt] = __builtin_amdgcn_wmma_f32_16x16x32_bf16(false, a, false, bf,
                                                        (short)0, acc[nt], false, false);
    }
  }
#pragma unroll
  for (int nt = 0; nt < 4; ++nt) {
    int c = nt * 16 + (lane & 15);
    float bias = b_out[c];
#pragma unroll
    for (int vv = 0; vv < 8; ++vv) {
      int mm = vv + ((lane >> 4) << 3);
      size_t row = mt * 16 + mm;
      out[row * RULE + c] = fmaxf(0.0f, acc[nt][vv] + bias);
    }
  }
}

extern "C" void kernel_launch(void* const* d_in, const int* in_sizes, int n_in,
                              void* d_out, int out_size, void* d_ws, size_t ws_size,
                              hipStream_t stream) {
  (void)in_sizes; (void)n_in; (void)out_size;
  const float* stim  = (const float*)d_in[0];
  const float* W_ih  = (const float*)d_in[1];
  const float* W_hh  = (const float*)d_in[2];
  const float* b_ih  = (const float*)d_in[3];
  const float* b_hh  = (const float*)d_in[4];
  const float* W_out = (const float*)d_in[5];
  const float* b_out = (const float*)d_in[6];

  float* out = (float*)d_out;                         // [B,T,64]
  float* hid = out + (size_t)BATCH * SEQ * RULE;      // [B,T,256]

  const bool use_ws = (d_ws != nullptr) && (ws_size >= (size_t)TOTAL_E * sizeof(__bf16));

  if (use_ws) {
    __bf16* ws      = (__bf16*)d_ws;
    __bf16* whh_fr  = ws;
    __bf16* wih_fr  = ws + WHH_E;
    __bf16* wout_fr = ws + WHH_E + WIH_E;
    preconvert_kernel<<<(TOTAL_E + 511) / 512, 512, 0, stream>>>(W_ih, W_hh, W_out, ws);
    gru_relu_kernel<true><<<BATCH / 16, 512, 0, stream>>>(stim, W_ih, W_hh, b_ih, b_hh,
                                                          whh_fr, wih_fr, hid);
    head_kernel<true><<<(BATCH * SEQ / 16) / 8, 256, 0, stream>>>(hid, W_out, b_out,
                                                                  wout_fr, out);
  } else {
    gru_relu_kernel<false><<<BATCH / 16, 512, 0, stream>>>(stim, W_ih, W_hh, b_ih, b_hh,
                                                           nullptr, nullptr, hid);
    head_kernel<false><<<(BATCH * SEQ / 16) / 8, 256, 0, stream>>>(hid, W_out, b_out,
                                                                   nullptr, out);
  }
}
